// EncoderDecoderAttention_91070486544497
// MI455X (gfx1250) — matrix operand order repaired and checked by reading, not compiled, not run on hardware
//
#include <hip/hip_runtime.h>
#include <hip/hip_bf16.h>

// ---------------------------------------------------------------------------
// EncoderDecoderAttention for MI455X (gfx1250, wave32, WMMA).
// B=2, S=2048, D=1024, H=16, DK=64. fp32 in/out, bf16 WMMA internally.
// Pipeline: 3x projection GEMM (fp32 -> bf16 headsplit) -> flash attention
//           (bf16 WMMA QK^T and P@V, fp32 online softmax, async K/V staging)
//           -> output GEMM (bf16 ctx x fp32 WO -> fp32 out).
// ---------------------------------------------------------------------------

constexpr int Bc  = 2;
constexpr int Sc  = 2048;
constexpr int Dc  = 1024;
constexpr int Hc  = 16;
constexpr int DKc = 64;

typedef __attribute__((ext_vector_type(16))) __bf16 v16bf;
typedef __attribute__((ext_vector_type(8)))  float  v8f;

// ---- helpers ---------------------------------------------------------------

__device__ inline __bf16 f2bf(float f) {
    // round-to-nearest-even fp32 -> bf16
    unsigned int u = __float_as_uint(f);
    u += 0x7FFFu + ((u >> 16) & 1u);
    union { unsigned short s; __bf16 b; } cvt;
    cvt.s = (unsigned short)(u >> 16);
    return cvt.b;
}
__device__ inline __bf16 to_bf(float f)  { return f2bf(f); }
__device__ inline __bf16 to_bf(__bf16 b) { return b; }

__device__ inline v8f wmma_bf16(v16bf a, v16bf b, v8f c) {
    // v_wmma_f32_16x16x32_bf16: (neg_a, A, neg_b, B, c_mod, C, reuse_a, reuse_b)
    return __builtin_amdgcn_wmma_f32_16x16x32_bf16(false, a, false, b,
                                                   (short)0, c, false, false);
}

// Async global->LDS copy (16B per lane), tracked by ASYNCcnt (cdna5 §10.7/§15.18).
// LDS offset = low 32 bits of the generic pointer (LDS aperture truncation).
__device__ inline void async_ld_b128(unsigned lds_off, const void* gptr) {
    asm volatile("global_load_async_to_lds_b128 %0, %1, off"
                 :: "v"(lds_off), "v"(gptr)
                 : "memory");
}
__device__ inline void wait_async0() {
    asm volatile("s_wait_asynccnt 0x0" ::: "memory");
}

// Row of a 16x16 C-fragment lives in a 16-lane half on wave32 -> reduce width 16.
__device__ inline float rowmax16(float v) {
    #pragma unroll
    for (int off = 8; off; off >>= 1) v = fmaxf(v, __shfl_xor(v, off, 16));
    return v;
}
__device__ inline float rowsum16(float v) {
    #pragma unroll
    for (int off = 8; off; off >>= 1) v += __shfl_xor(v, off, 16);
    return v;
}

// A-fragment (16x32 bf16) element j -> K offset, contiguous order (k = 16*hi + j).
__device__ inline int a_koff(int j, int hi) {
    return j + (hi << 4);
}

// ---- generic 64x64-tile WMMA GEMM ------------------------------------------
// C[M,N] = A[M,K] * Bm[K,N].  AT = float (convert) or __bf16 (pass-through).
// HEADSPLIT: store bf16 to [B,H,S,DK]; else store fp32 row-major [M,N].
template <typename AT, bool HEADSPLIT>
__global__ __launch_bounds__(128)
void gemm64x64_wmma(const AT* __restrict__ A, const float* __restrict__ Bm,
                    void* __restrict__ OutP, int M, int N, int K) {
    __shared__ __bf16 As[64][34];   // 64 rows x 32 cols (+pad)
    __shared__ __bf16 Bs[32][66];   // 32 rows x 64 cols (+pad)

    const int t    = threadIdx.x;
    const int lane = t & 31;
    const int w    = t >> 5;          // 4 waves, wave w -> rows [w*16, w*16+16)
    const int n    = lane & 15;
    const int hi   = lane >> 4;
    const int m0   = blockIdx.x * 64;
    const int n0   = blockIdx.y * 64;

    v8f acc[4];
    #pragma unroll
    for (int nc = 0; nc < 4; ++nc) acc[nc] = {};

    for (int k0 = 0; k0 < K; k0 += 32) {
        __syncthreads();
        // cooperative stage: A tile 64x32, B tile 32x64 (fp32->bf16 convert)
        for (int e = t; e < 64 * 32; e += 128) {
            int r = e >> 5, c = e & 31;
            As[r][c] = to_bf(A[(size_t)(m0 + r) * K + k0 + c]);
        }
        for (int e = t; e < 32 * 64; e += 128) {
            int r = e >> 6, c = e & 63;
            Bs[r][c] = f2bf(Bm[(size_t)(k0 + r) * N + n0 + c]);
        }
        // prefetch next k-tile into cache (global_prefetch_b8)
        if (k0 + 32 < K) {
            __builtin_prefetch(&A[(size_t)(m0 + (t >> 1)) * K + k0 + 32], 0, 1);
            __builtin_prefetch(&Bm[(size_t)(k0 + 32 + (t >> 2)) * N + n0], 0, 1);
        }
        __syncthreads();

        v16bf a;
        const int arow = w * 16 + n;
        #pragma unroll
        for (int j = 0; j < 16; ++j) a[j] = As[arow][a_koff(j, hi)];

        #pragma unroll
        for (int nc = 0; nc < 4; ++nc) {
            v16bf b;
            #pragma unroll
            for (int j = 0; j < 16; ++j) b[j] = Bs[j + hi * 16][nc * 16 + n];
            acc[nc] = wmma_bf16(a, b, acc[nc]);
        }
    }

    // store C fragments: row = i + 8*hi, col = lane&15
    #pragma unroll
    for (int nc = 0; nc < 4; ++nc) {
        #pragma unroll
        for (int i = 0; i < 8; ++i) {
            int r    = m0 + w * 16 + i + 8 * hi;
            int ccol = n0 + nc * 16 + n;
            if (HEADSPLIT) {
                int bi = r / Sc, s = r % Sc;
                int hh = ccol >> 6, dk = ccol & 63;
                ((__bf16*)OutP)[(((size_t)bi * Hc + hh) * Sc + s) * DKc + dk] =
                    f2bf(acc[nc][i]);
            } else {
                ((float*)OutP)[(size_t)r * N + ccol] = acc[nc][i];
            }
        }
    }
}

// ---- flash attention kernel ------------------------------------------------
// grid = (S/64 query tiles, B*H). block = 128 (4 waves x 16 queries).
__global__ __launch_bounds__(128)
void attn_fwd(const __bf16* __restrict__ Qb, const __bf16* __restrict__ Kb,
              const __bf16* __restrict__ Vb, const int* __restrict__ maskp,
              __bf16* __restrict__ Ctx) {
    __shared__ __bf16 Ks[32][64];        // 32 keys x 64 dk (flat: async target)
    __shared__ __bf16 Vs[32][64];
    __shared__ __bf16 Ps[4][16][34];     // per-wave P tile, 16q x 32k (+pad)

    const int t    = threadIdx.x;
    const int lane = t & 31;
    const int w    = t >> 5;
    const int n    = lane & 15;
    const int hi   = lane >> 4;
    const int bh   = blockIdx.y;
    const int b    = bh >> 4;            // H = 16
    const int h    = bh & 15;
    const int q0   = blockIdx.x * 64 + w * 16;

    const unsigned ldsK = (unsigned)(size_t)&Ks[0][0];
    const unsigned ldsV = (unsigned)(size_t)&Vs[0][0];

    // Q fragments (16 queries x 64 dk = two 16x32 A-fragments), loaded once.
    const __bf16* qrow = Qb + ((size_t)bh * Sc + q0 + n) * DKc;
    v16bf aq0, aq1;
    #pragma unroll
    for (int j = 0; j < 16; ++j) {
        int ko = a_koff(j, hi);
        aq0[j] = qrow[ko];
        aq1[j] = qrow[32 + ko];
    }

    float mi[8], li[8];
    bool  rmask[8];
    #pragma unroll
    for (int i = 0; i < 8; ++i) {
        int q   = q0 + i + 8 * hi;
        rmask[i] = maskp[b * Sc + q] != 0;  // padding mask kills whole query row
        mi[i] = -3.0e38f;
        li[i] = 0.0f;
    }
    v8f O[4];
    #pragma unroll
    for (int nc = 0; nc < 4; ++nc) O[nc] = {};

    const int ktiles = 2 * blockIdx.x + 2;  // causal: only keys <= q0+63 matter
    for (int kt = 0; kt < ktiles; ++kt) {
        __syncthreads();
        {   // async stage of 32x64 bf16 K and V tiles (4KB each, contiguous):
            // 256 x 16B chunks / 128 threads = 2 per thread per tile.
            const char* gk = (const char*)(Kb + ((size_t)bh * Sc + kt * 32) * DKc);
            const char* gv = (const char*)(Vb + ((size_t)bh * Sc + kt * 32) * DKc);
            #pragma unroll
            for (int r = 0; r < 2; ++r) {
                int e = (t + r * 128) * 16;
                async_ld_b128(ldsK + e, gk + e);
                async_ld_b128(ldsV + e, gv + e);
            }
            wait_async0();
        }
        __syncthreads();

        // S = Q * K^T : two 16-key subtiles, each accumulated over dk in 2 steps
        v8f s0 = {}, s1 = {};
        {
            v16bf bk;
            #pragma unroll
            for (int j = 0; j < 16; ++j) bk[j] = Ks[n][j + hi * 16];
            s0 = wmma_bf16(aq0, bk, s0);
            #pragma unroll
            for (int j = 0; j < 16; ++j) bk[j] = Ks[n][32 + j + hi * 16];
            s0 = wmma_bf16(aq1, bk, s0);
            #pragma unroll
            for (int j = 0; j < 16; ++j) bk[j] = Ks[16 + n][j + hi * 16];
            s1 = wmma_bf16(aq0, bk, s1);
            #pragma unroll
            for (int j = 0; j < 16; ++j) bk[j] = Ks[16 + n][32 + j + hi * 16];
            s1 = wmma_bf16(aq1, bk, s1);
        }

        // online softmax over this 32-key tile
        float alpha[8];
        #pragma unroll
        for (int i = 0; i < 8; ++i) {
            int   q   = q0 + i + 8 * hi;
            int   k0i = kt * 32 + n;
            int   k1i = kt * 32 + 16 + n;
            float v0 = (rmask[i] || k0i > q) ? -1e9f : s0[i] * 0.125f;
            float v1 = (rmask[i] || k1i > q) ? -1e9f : s1[i] * 0.125f;
            float rm = rowmax16(fmaxf(v0, v1));
            float nm = fmaxf(mi[i], rm);
            float e0 = __expf(v0 - nm);
            float e1 = __expf(v1 - nm);
            float rs = rowsum16(e0 + e1);
            alpha[i] = __expf(mi[i] - nm);
            li[i]    = alpha[i] * li[i] + rs;
            mi[i]    = nm;
            // stash P in per-wave LDS (C-layout -> plain [q][k]) as bf16
            Ps[w][i + 8 * hi][n]      = f2bf(e0);
            Ps[w][i + 8 * hi][16 + n] = f2bf(e1);
        }
        #pragma unroll
        for (int nc = 0; nc < 4; ++nc) {
            #pragma unroll
            for (int i = 0; i < 8; ++i) O[nc][i] *= alpha[i];
        }

        __builtin_amdgcn_wave_barrier();  // order Ps stores vs A-frag gathers

        // O += P * V : re-read P as a 16x32 A-fragment, V tile as 4 B-fragments
        v16bf ap;
        #pragma unroll
        for (int j = 0; j < 16; ++j) ap[j] = Ps[w][n][a_koff(j, hi)];
        #pragma unroll
        for (int nc = 0; nc < 4; ++nc) {
            v16bf bv;
            #pragma unroll
            for (int j = 0; j < 16; ++j) bv[j] = Vs[j + hi * 16][nc * 16 + n];
            O[nc] = wmma_bf16(ap, bv, O[nc]);
        }
    }

    // epilogue: O / l, store ctx as bf16 [B,S,D] (D = h*64 + dk)
    #pragma unroll
    for (int i = 0; i < 8; ++i) {
        float inv = 1.0f / li[i];
        int   q   = q0 + i + 8 * hi;
        #pragma unroll
        for (int nc = 0; nc < 4; ++nc) {
            Ctx[((size_t)b * Sc + q) * Dc + h * DKc + nc * 16 + n] =
                f2bf(O[nc][i] * inv);
        }
    }
}

// ---- launch ----------------------------------------------------------------

extern "C" void kernel_launch(void* const* d_in, const int* in_sizes, int n_in,
                              void* d_out, int out_size, void* d_ws, size_t ws_size,
                              hipStream_t stream) {
    (void)in_sizes; (void)n_in; (void)out_size; (void)ws_size;

    const float* x   = (const float*)d_in[0];
    const float* enc = (const float*)d_in[1];
    const int*   msk = (const int*)d_in[2];
    const float* WQ  = (const float*)d_in[3];
    const float* WK  = (const float*)d_in[4];
    const float* WV  = (const float*)d_in[5];
    const float* WO  = (const float*)d_in[6];

    const size_t PER = (size_t)Bc * Hc * Sc * DKc;   // 4Mi bf16 per buffer
    __bf16* Qb  = (__bf16*)d_ws;
    __bf16* Kb  = Qb + PER;
    __bf16* Vb  = Kb + PER;
    __bf16* Ctx = Vb + PER;                          // [B*S, D] bf16

    const int M = Bc * Sc, N = Dc, K = Dc;
    dim3 gg(M / 64, N / 64);                          // (64, 16)

    gemm64x64_wmma<float, true ><<<gg, 128, 0, stream>>>(x,   WQ, Qb,  M, N, K);
    gemm64x64_wmma<float, true ><<<gg, 128, 0, stream>>>(enc, WK, Kb,  M, N, K);
    gemm64x64_wmma<float, true ><<<gg, 128, 0, stream>>>(enc, WV, Vb,  M, N, K);

    dim3 ga(Sc / 64, Bc * Hc);                        // (32, 32)
    attn_fwd<<<ga, 128, 0, stream>>>(Qb, Kb, Vb, msk, Ctx);

    gemm64x64_wmma<__bf16, false><<<gg, 128, 0, stream>>>(Ctx, WO, d_out, M, N, K);
}
